// SparseStructuredAttention_74285754352206
// MI455X (gfx1250) — matrix operand order repaired and checked by reading, not compiled, not run on hardware
//
#include <hip/hip_runtime.h>

typedef __attribute__((ext_vector_type(16))) __bf16       v16bf;
typedef __attribute__((ext_vector_type(8)))  __bf16       v8bf;
typedef __attribute__((ext_vector_type(8)))  float        v8f;
typedef __attribute__((ext_vector_type(4)))  unsigned int u32x4;
typedef __attribute__((ext_vector_type(4)))  int          i32x4;
typedef __attribute__((ext_vector_type(8)))  int          i32x8;

#define B_    2
#define S_    2048
#define HID_  1024
#define NH_   16
#define HD_   64
#define HWIN_ 64
#define MTOT_ (B_ * S_)
#define NT_   (S_ / 32)     // key tiles of 32

#if __has_builtin(__builtin_amdgcn_tensor_load_to_lds) && \
    __has_builtin(__builtin_amdgcn_s_wait_tensorcnt)
#define USE_TDM 1
#else
#define USE_TDM 0
#endif

static __device__ __forceinline__ __bf16 f2bf(float f) { return (__bf16)f; }

static __device__ __forceinline__ v16bf cat8(v8bf lo, v8bf hi) {
    return __builtin_shufflevector(lo, hi, 0, 1, 2, 3, 4, 5, 6, 7,
                                   8, 9, 10, 11, 12, 13, 14, 15);
}

#if USE_TDM
// ---------------------------------------------------------------------------
// Issue one TDM 2-D tile load: global (elems of 2B, row stride stride0 elems,
// tile tile0 x tile1) -> LDS at lds_off, row-major tile0-contiguous.
// D# packing per CDNA5 ISA 8.3/8.4 (group0: count/lds/global/type,
// group1: data_size=2B, dims, tile dims, 48-bit stride).
// ---------------------------------------------------------------------------
static __device__ __forceinline__ void tdm_load_2d(
    unsigned lds_off, unsigned long long gaddr,
    unsigned td0, unsigned td1, unsigned tile0, unsigned tile1,
    unsigned long long stride0)
{
    u32x4 g0;
    g0[0] = 1u;                                               // count=1, user D#
    g0[1] = lds_off;                                          // LDS byte address
    g0[2] = (unsigned)(gaddr & 0xFFFFFFFFull);                // global_addr[31:0]
    g0[3] = (unsigned)((gaddr >> 32) & 0x1FFFFFFull) | (2u << 30);  // [56:32]|type=2

    i32x8 g1;
    g1[0] = (int)(1u << 16);                                  // data_size=1 -> 2B
    g1[1] = (int)((td0 & 0xFFFFu) << 16);                     // tensor_dim0[15:0]
    g1[2] = (int)(((td0 >> 16) & 0xFFFFu) | ((td1 & 0xFFFFu) << 16));
    g1[3] = (int)(((td1 >> 16) & 0xFFFFu) | ((tile0 & 0xFFFFu) << 16));
    g1[4] = (int)(tile1 & 0xFFFFu);                           // tile_dim1, tile_dim2=0
    g1[5] = (int)(stride0 & 0xFFFFFFFFull);                   // dim0 stride [31:0]
    g1[6] = (int)((stride0 >> 32) & 0xFFFFull);               // dim0 stride [47:32]
    g1[7] = 0;

    i32x4 z4 = {0, 0, 0, 0};
#if defined(__clang_major__) && (__clang_major__ >= 23)
    i32x8 z8 = {0, 0, 0, 0, 0, 0, 0, 0};
    __builtin_amdgcn_tensor_load_to_lds(g0, g1, z4, z4, z8, 0);
#else
    __builtin_amdgcn_tensor_load_to_lds(g0, g1, z4, z4, 0);
#endif
}
#endif

// ---------------------------------------------------------------------------
// fp32 -> bf16 bulk conversion (8 elements/thread).
// ---------------------------------------------------------------------------
__global__ __launch_bounds__(256) void cvt_f32_to_bf16(
    const float* __restrict__ in, __bf16* __restrict__ out, int n)
{
    const int i = (blockIdx.x * 256 + threadIdx.x) * 8;
    if (i + 8 > n) return;
    const float4 a = *(const float4*)(in + i);
    const float4 b = *(const float4*)(in + i + 4);
    v8bf o;
    o[0] = f2bf(a.x); o[1] = f2bf(a.y); o[2] = f2bf(a.z); o[3] = f2bf(a.w);
    o[4] = f2bf(b.x); o[5] = f2bf(b.y); o[6] = f2bf(b.z); o[7] = f2bf(b.w);
    *(v8bf*)(out + i) = o;
}

// ---------------------------------------------------------------------------
// GEMM: Y[M,N] = (X[M,K] @ W[N,K]^T + bias[N]) * out_scale, bf16 WMMA.
// One wave32 computes a 32x64 tile (8 WMMA per 32-wide k-step).
// OUT_MODE: 0 = f32 row-major, 1 = bf16 row-major, 2 = bf16 transposed [HID][S]
// ---------------------------------------------------------------------------
template <int OUT_MODE>
__global__ __launch_bounds__(32) void gemm32x64(
    const __bf16* __restrict__ X, const __bf16* __restrict__ W,
    const float* __restrict__ bias, void* __restrict__ Yv, float out_scale)
{
    const int l  = threadIdx.x;
    const int g  = l >> 4;
    const int lm = l & 15;
    const int m0 = blockIdx.y * 32;
    const int n0 = blockIdx.x * 64;

    v8f acc[2][4];
    #pragma unroll
    for (int mi = 0; mi < 2; ++mi)
        #pragma unroll
        for (int nb = 0; nb < 4; ++nb)
            #pragma unroll
            for (int e = 0; e < 8; ++e) acc[mi][nb][e] = 0.0f;

    const __bf16* xrow0 = X + (size_t)(m0 + lm) * HID_;
    const __bf16* xrow1 = X + (size_t)(m0 + 16 + lm) * HID_;

    for (int k0 = 0; k0 < HID_; k0 += 32) {
        const v16bf a0 = cat8(*(const v8bf*)(xrow0 + k0 + g * 8),
                              *(const v8bf*)(xrow0 + k0 + 16 + g * 8));
        const v16bf a1 = cat8(*(const v8bf*)(xrow1 + k0 + g * 8),
                              *(const v8bf*)(xrow1 + k0 + 16 + g * 8));
        if (k0 + 32 < HID_) {
            __builtin_prefetch((const void*)(xrow0 + k0 + 32), 0, 3);
            __builtin_prefetch((const void*)(xrow1 + k0 + 32), 0, 3);
        }
        #pragma unroll
        for (int nb = 0; nb < 4; ++nb) {
            const __bf16* wrow = W + (size_t)(n0 + nb * 16 + lm) * HID_ + k0;
            const v16bf bm = cat8(*(const v8bf*)(wrow + g * 8),
                                  *(const v8bf*)(wrow + 16 + g * 8));
            acc[0][nb] = __builtin_amdgcn_wmma_f32_16x16x32_bf16(
                false, a0, false, bm, (short)0, acc[0][nb], false, false);
            acc[1][nb] = __builtin_amdgcn_wmma_f32_16x16x32_bf16(
                false, a1, false, bm, (short)0, acc[1][nb], false, false);
        }
    }

    #pragma unroll
    for (int nb = 0; nb < 4; ++nb) {
        const int col = n0 + nb * 16 + lm;
        const float bv = bias[col];
        #pragma unroll
        for (int mi = 0; mi < 2; ++mi) {
            if (OUT_MODE == 2) {
                const int bidx = m0 / S_;
                const int s0   = m0 - bidx * S_ + mi * 16 + g * 8;
                v8bf o;
                #pragma unroll
                for (int r = 0; r < 8; ++r)
                    o[r] = f2bf((acc[mi][nb][r] + bv) * out_scale);
                *(v8bf*)((__bf16*)Yv + ((size_t)bidx * HID_ + col) * S_ + s0) = o;
            } else {
                #pragma unroll
                for (int r = 0; r < 8; ++r) {
                    const int row = m0 + mi * 16 + g * 8 + r;
                    const float val = (acc[mi][nb][r] + bv) * out_scale;
                    if (OUT_MODE == 0)
                        ((float*)Yv)[(size_t)row * HID_ + col] = val;
                    else
                        ((__bf16*)Yv)[(size_t)row * HID_ + col] = f2bf(val);
                }
            }
        }
    }
}

// ---------------------------------------------------------------------------
// Per-batch key-tile (32 keys) "contains global token" flags.
// ---------------------------------------------------------------------------
__global__ void ktile_flags(const int* __restrict__ ig, int* __restrict__ flags)
{
    const int b = blockIdx.x;
    const int t = threadIdx.x;
    const int* p = ig + b * S_ + t * 32;
    int any = 0;
    #pragma unroll
    for (int i = 0; i < 32; ++i) any |= p[i];
    flags[b * 64 + t] = (any != 0);
}

// ---------------------------------------------------------------------------
// Sparse flash attention. K/V tiles staged to LDS by the Tensor Data Mover,
// double-buffered across the (sparse) tile sequence; WMMA operands read from
// LDS. One wave32 per (16-row query tile, head, batch).
// ---------------------------------------------------------------------------
__global__ __launch_bounds__(32) void sattn(
    const __bf16* __restrict__ Q, const __bf16* __restrict__ K,
    const __bf16* __restrict__ Vt, const int* __restrict__ ig,
    const int* __restrict__ ktf, __bf16* __restrict__ AO)
{
    __shared__ __attribute__((aligned(16))) __bf16 kbuf[2][32 * 64]; // [key][d]
    __shared__ __attribute__((aligned(16))) __bf16 vbuf[2][64 * 32]; // [d][key]
    __shared__ float sc[16][32];
    __shared__ __attribute__((aligned(16))) __bf16 pt[16][32];
    __shared__ float alpha_s[16];
    __shared__ float lsum_s[16];
    __shared__ int   kf_s[32];

    const int l  = threadIdx.x;
    const int g  = l >> 4;
    const int lm = l & 15;
    const int qt = blockIdx.x, h = blockIdx.y, b = blockIdx.z;
    const int q0 = qt * 16;

    // Q fragments (scale folded into the Q projection).
    const __bf16* qrow = Q + (size_t)(b * S_ + q0 + lm) * HID_ + h * HD_;
    const v16bf qa0 = cat8(*(const v8bf*)(qrow + g * 8),
                           *(const v8bf*)(qrow + 16 + g * 8));
    const v16bf qa1 = cat8(*(const v8bf*)(qrow + 32 + g * 8),
                           *(const v8bf*)(qrow + 48 + g * 8));

    const int  qg     = ig[b * S_ + q0 + lm] != 0;
    const bool any_qg = (__ballot(qg) != 0ull);

    auto tile_needed = [&](int t) -> bool {
        const int key0 = t * 32;
        const bool in_window = (key0 + 31 >= q0 - HWIN_) && (key0 <= q0 + 15 + HWIN_);
        return any_qg || (ktf[b * 64 + t] != 0) || in_window;
    };
    auto next_tile = [&](int t) -> int {
        while (t < NT_ && !tile_needed(t)) ++t;
        return t;
    };

#if USE_TDM
    auto issue_tile = [&](int t, int p) {
        const int key0 = t * 32;
        // K tile: 32 key rows x 64 contiguous dims, row stride HID_ elems.
        tdm_load_2d((unsigned)(size_t)&kbuf[p][0],
                    (unsigned long long)(size_t)(K + (size_t)(b * S_ + key0) * HID_ + h * HD_),
                    64u, 32u, 64u, 32u, (unsigned long long)HID_);
        // V tile (from transposed V): 64 d rows x 32 contiguous keys, stride S_.
        tdm_load_2d((unsigned)(size_t)&vbuf[p][0],
                    (unsigned long long)(size_t)(Vt + ((size_t)b * HID_ + h * HD_) * S_ + key0),
                    32u, 64u, 32u, 64u, (unsigned long long)S_);
    };
#else
    auto copy_tile = [&](int t, int p) {
        const int key0 = t * 32;
        #pragma unroll
        for (int c = 0; c < 8; ++c) {            // 256 x 8-elem chunks, 8 per lane
            const int o   = (c * 32 + l) * 8;    // K tile: [row=o/64][col=o%64]
            const int row = o >> 6, col = o & 63;
            *(v8bf*)&kbuf[p][o] =
                *(const v8bf*)(K + (size_t)(b * S_ + key0 + row) * HID_ + h * HD_ + col);
            const int rv = o >> 5, cv = o & 31;  // V tile: [d=o/32][key=o%32]
            *(v8bf*)&vbuf[p][o] =
                *(const v8bf*)(Vt + ((size_t)b * HID_ + h * HD_ + rv) * S_ + key0 + cv);
        }
    };
#endif

    v8f o[4];
    #pragma unroll
    for (int nb = 0; nb < 4; ++nb)
        #pragma unroll
        for (int e = 0; e < 8; ++e) o[nb][e] = 0.0f;

    float m_run = -1e30f, l_run = 0.0f;

    int cur = next_tile(0);
    int parity = 0;
#if USE_TDM
    if (cur < NT_) issue_tile(cur, 0);
#endif

    while (cur < NT_) {
        const int key0 = cur * 32;
        const int nxt  = next_tile(cur + 1);

#if USE_TDM
        if (nxt < NT_) {
            issue_tile(nxt, parity ^ 1);                 // prefetch next tile pair
            __builtin_amdgcn_s_wait_tensorcnt((short)2); // current K+V landed
        } else {
            __builtin_amdgcn_s_wait_tensorcnt((short)0);
        }
        asm volatile("" ::: "memory");
#else
        copy_tile(cur, parity);
        __syncthreads();
#endif

        kf_s[l] = ig[b * S_ + key0 + l];

        // ---- scores: 16x32 = Q(16x64) @ K^T, operands from LDS ----
        v8f c0, c1;
        #pragma unroll
        for (int e = 0; e < 8; ++e) { c0[e] = 0.0f; c1[e] = 0.0f; }

        #pragma unroll
        for (int kh = 0; kh < 2; ++kh) {
            const __bf16* krow = &kbuf[parity][(kh * 16 + lm) * 64];
            const v16bf kb0 = cat8(*(const v8bf*)(krow + g * 8),
                                   *(const v8bf*)(krow + 16 + g * 8));
            const v16bf kb1 = cat8(*(const v8bf*)(krow + 32 + g * 8),
                                   *(const v8bf*)(krow + 48 + g * 8));
            if (kh == 0) {
                c0 = __builtin_amdgcn_wmma_f32_16x16x32_bf16(false, qa0, false, kb0, (short)0, c0, false, false);
                c0 = __builtin_amdgcn_wmma_f32_16x16x32_bf16(false, qa1, false, kb1, (short)0, c0, false, false);
            } else {
                c1 = __builtin_amdgcn_wmma_f32_16x16x32_bf16(false, qa0, false, kb0, (short)0, c1, false, false);
                c1 = __builtin_amdgcn_wmma_f32_16x16x32_bf16(false, qa1, false, kb1, (short)0, c1, false, false);
            }
        }

        #pragma unroll
        for (int r = 0; r < 8; ++r) {
            sc[g * 8 + r][lm]      = c0[r];
            sc[g * 8 + r][16 + lm] = c1[r];
        }
        __syncthreads();

        // ---- online softmax: lane = (row lm, column half g) ----
        const int i_glob = q0 + lm;
        float sv[16];
        float tmax = -1e30f;
        #pragma unroll
        for (int c = 0; c < 16; ++c) {
            const int j = key0 + g * 16 + c;
            const bool ok = qg || (kf_s[g * 16 + c] != 0) ||
                            ((i_glob - j) <= HWIN_ && (j - i_glob) <= HWIN_);
            const float s = ok ? sc[lm][g * 16 + c] : -1e30f;
            sv[c] = s;
            tmax  = fmaxf(tmax, s);
        }
        tmax = fmaxf(tmax, __shfl_xor(tmax, 16, 32));
        const float m_new = fmaxf(m_run, tmax);
        const float alpha = __expf(m_run - m_new);
        float rs = 0.0f;
        #pragma unroll
        for (int c = 0; c < 16; ++c) {
            const float p = __expf(sv[c] - m_new);
            rs += p;
            pt[lm][g * 16 + c] = f2bf(p);
        }
        rs += __shfl_xor(rs, 16, 32);
        l_run = alpha * l_run + rs;
        m_run = m_new;
        if (g == 0) alpha_s[lm] = alpha;
        __syncthreads();

        // ---- rescale accumulator, then O += P @ V_tile (V from LDS) ----
        #pragma unroll
        for (int nb = 0; nb < 4; ++nb)
            #pragma unroll
            for (int r = 0; r < 8; ++r) o[nb][r] *= alpha_s[g * 8 + r];

        const v16bf pa = cat8(*(const v8bf*)(&pt[lm][g * 8]),
                              *(const v8bf*)(&pt[lm][16 + g * 8]));

        #pragma unroll
        for (int nb = 0; nb < 4; ++nb) {
            const __bf16* vcol = &vbuf[parity][(nb * 16 + lm) * 32];
            const v16bf vb = cat8(*(const v8bf*)(vcol + g * 8),
                                  *(const v8bf*)(vcol + 16 + g * 8));
            o[nb] = __builtin_amdgcn_wmma_f32_16x16x32_bf16(false, pa, false, vb, (short)0, o[nb], false, false);
        }
        __syncthreads();

        cur    = nxt;
        parity ^= 1;
    }

    // ---- finalize ----
    if (g == 0) lsum_s[lm] = l_run;
    __syncthreads();
    #pragma unroll
    for (int nb = 0; nb < 4; ++nb)
        #pragma unroll
        for (int r = 0; r < 8; ++r) {
            const int row = g * 8 + r;
            const float inv = 1.0f / lsum_s[row];
            AO[(size_t)(b * S_ + q0 + row) * HID_ + h * HD_ + nb * 16 + lm] =
                f2bf(o[nb][r] * inv);
        }
}

// ---------------------------------------------------------------------------
extern "C" void kernel_launch(void* const* d_in, const int* in_sizes, int n_in,
                              void* d_out, int out_size, void* d_ws, size_t ws_size,
                              hipStream_t stream) {
    const float* x  = (const float*)d_in[0];
    const int*   ig = (const int*)  d_in[1];
    const float* wq = (const float*)d_in[2];
    const float* bq = (const float*)d_in[3];
    const float* wk = (const float*)d_in[4];
    const float* bk = (const float*)d_in[5];
    const float* wv = (const float*)d_in[6];
    const float* bv = (const float*)d_in[7];
    const float* wo = (const float*)d_in[8];
    const float* bo = (const float*)d_in[9];
    float* out = (float*)d_out;

    const size_t MAT = (size_t)MTOT_ * HID_;
    const size_t WEL = (size_t)HID_ * HID_;

    __bf16* xb  = (__bf16*)d_ws;
    __bf16* wqb = xb + MAT;
    __bf16* wkb = wqb + WEL;
    __bf16* wvb = wkb + WEL;
    __bf16* wob = wvb + WEL;
    __bf16* qbf = wob + WEL;
    __bf16* kbf = qbf + MAT;
    __bf16* vtb = kbf + MAT;          // V transposed per batch [HID][S]
    __bf16* aob = vtb + MAT;
    int*    kf  = (int*)(aob + MAT);

    cvt_f32_to_bf16<<<(int)(MAT / (8 * 256)), 256, 0, stream>>>(x, xb, (int)MAT);
    cvt_f32_to_bf16<<<(int)(WEL / (8 * 256)), 256, 0, stream>>>(wq, wqb, (int)WEL);
    cvt_f32_to_bf16<<<(int)(WEL / (8 * 256)), 256, 0, stream>>>(wk, wkb, (int)WEL);
    cvt_f32_to_bf16<<<(int)(WEL / (8 * 256)), 256, 0, stream>>>(wv, wvb, (int)WEL);
    cvt_f32_to_bf16<<<(int)(WEL / (8 * 256)), 256, 0, stream>>>(wo, wob, (int)WEL);

    const float SCALE = 0.125f;
    dim3 gemm_grid(HID_ / 64, MTOT_ / 32);
    gemm32x64<1><<<gemm_grid, 32, 0, stream>>>(xb, wqb, bq, qbf, SCALE);
    gemm32x64<1><<<gemm_grid, 32, 0, stream>>>(xb, wkb, bk, kbf, 1.0f);
    gemm32x64<2><<<gemm_grid, 32, 0, stream>>>(xb, wvb, bv, vtb, 1.0f);

    ktile_flags<<<B_, 64, 0, stream>>>(ig, kf);
    dim3 attn_grid(S_ / 16, NH_, B_);
    sattn<<<attn_grid, 32, 0, stream>>>(qbf, kbf, vtb, ig, kf, aob);

    gemm32x64<0><<<gemm_grid, 32, 0, stream>>>(aob, wob, bo, out, 1.0f);
}
